// MinGRU_38981123178816
// MI455X (gfx1250) — matrix operand order repaired and checked by reading, not compiled, not run on hardware
//
#include <hip/hip_runtime.h>
#include <hip/hip_bf16.h>
#include <math.h>

// ---------------------------------------------------------------------------
// MinGRU forward for MI455X (gfx1250, wave32, WMMA)
//   k   = x @ Wz^T + bz          (B,S,H)
//   pre = x @ Wh^T + bh          (B,S,H)
//   log-space recurrence:  log_h_t = logaddexp(log(1-z_t) + log_h_{t-1},
//                                              log z_t + log g(pre_t))
//   h = exp(log_h)
// GEMMs: bf16x3 split-precision emulation of f32 (A~=Ahi+Alo, B~=Bhi+Blo,
// A.B ~= AhiBhi + AhiBlo + AloBhi) on v_wmma_f32_16x16x32_bf16.
// The hi/lo split is precomputed in global memory once, so the GEMM hot loop
// is pure: async global->LDS (ASYNCcnt, double buffered) -> ds_load_b128
// fragments -> WMMA. No conversion VALU in the loop.
// ---------------------------------------------------------------------------

#if defined(__has_builtin)
#  if __has_builtin(__builtin_amdgcn_global_load_async_to_lds_b128) && \
      __has_builtin(__builtin_amdgcn_s_wait_asynccnt)
#    define USE_ASYNC_LDS 1
#  endif
#endif
#ifndef USE_ASYNC_LDS
#  define USE_ASYNC_LDS 0
#endif

typedef __attribute__((ext_vector_type(16))) __bf16 v16bf;
typedef __attribute__((ext_vector_type(8)))  __bf16 v8bf;
typedef __attribute__((ext_vector_type(4)))  __bf16 v4bf;
typedef __attribute__((ext_vector_type(8)))  float  v8f;
typedef int v4i __attribute__((vector_size(16)));   // matches async builtin proto

static constexpr int BATCH = 8;
static constexpr int SEQ   = 4096;
static constexpr int DIN   = 1024;   // K
static constexpr int DH    = 1024;   // N
static constexpr int MTOT  = BATCH * SEQ;   // 32768
static constexpr int KTOT  = DIN;
static constexpr int NTOT  = DH;

static constexpr int BM = 128;
static constexpr int BN = 128;
static constexpr int BK = 32;
static constexpr int WAVES_M = 2;
static constexpr int WAVES_N = 4;
static constexpr int WM = BM / WAVES_M;   // 64
static constexpr int WN = BN / WAVES_N;   // 32
static constexpr int MT = WM / 16;        // 4 m-tiles per wave
static constexpr int NT = WN / 16;        // 2 n-tiles per wave
static constexpr int NK = KTOT / BK;      // 32 K-steps

// ---------------------------------------------------------------------------
// Elementwise split: hi = bf16(x) (RTNE), lo = bf16(x - hi). 4 elems/thread.
// ---------------------------------------------------------------------------
__global__ __launch_bounds__(256)
void split_bf16(const float* __restrict__ src,
                __bf16* __restrict__ hi,
                __bf16* __restrict__ lo)
{
    const int i = blockIdx.x * blockDim.x + threadIdx.x;   // float4 index
    const float4 f = ((const float4*)src)[i];
    const float v[4] = {f.x, f.y, f.z, f.w};
    v4bf h4, l4;
    #pragma unroll
    for (int c = 0; c < 4; ++c) {
        const __bf16 h = (__bf16)v[c];
        h4[c] = h;
        l4[c] = (__bf16)(v[c] - (float)h);
    }
    ((v4bf*)hi)[i] = h4;
    ((v4bf*)lo)[i] = l4;
}

// ---------------------------------------------------------------------------
// GEMM: out[m,n] = sum_k X[m,k] * W[n,k] + bias[n]
// A/B given as pre-split hi/lo bf16, (M,K) and (N,K) row-major.
// ---------------------------------------------------------------------------
__global__ __launch_bounds__(256, 1)
void mingru_gemm_bf16x3(const __bf16* __restrict__ Ahi,
                        const __bf16* __restrict__ Alo,
                        const __bf16* __restrict__ Bhi,
                        const __bf16* __restrict__ Blo,
                        const float*  __restrict__ bias,
                        float* __restrict__ out)
{
    __shared__ alignas(16) __bf16 sAhi[2][BM * BK];   // 2 x 8 KB
    __shared__ alignas(16) __bf16 sAlo[2][BM * BK];
    __shared__ alignas(16) __bf16 sBhi[2][BN * BK];
    __shared__ alignas(16) __bf16 sBlo[2][BN * BK];   // total 64 KB

    const int tid   = threadIdx.x;
    const int lane  = tid & 31;
    const int wid   = tid >> 5;          // 0..7
    const int waveM = wid >> 2;          // 0..1
    const int waveN = wid & 3;           // 0..3
    const int l16   = lane & 15;
    const int lg    = lane >> 4;         // 0/1 (lane group)

    const int mBlock = blockIdx.x * BM;
    const int nBlock = blockIdx.y * BN;

    // staging: each bf16 tile = 128 rows x 32 bf16 = 512 x 16B; 2 chunks/thread
    const int ldRow = tid >> 2;          // 0..63 (+64 for p=1)
    const int ldCol = (tid & 3) << 3;    // bf16 column 0,8,16,24

    auto issue_tile = [&](int kk, int buf) {
        #pragma unroll
        for (int p = 0; p < 2; ++p) {
            const int row = ldRow + p * 64;
            const size_t gAoff = (size_t)(mBlock + row) * KTOT + kk + ldCol;
            const size_t gBoff = (size_t)(nBlock + row) * KTOT + kk + ldCol;
            const int    loff  = row * BK + ldCol;
#if USE_ASYNC_LDS
            __builtin_amdgcn_global_load_async_to_lds_b128(
                (__attribute__((address_space(1))) v4i*)(Ahi + gAoff),
                (__attribute__((address_space(3))) v4i*)(&sAhi[buf][loff]), 0, 0);
            __builtin_amdgcn_global_load_async_to_lds_b128(
                (__attribute__((address_space(1))) v4i*)(Alo + gAoff),
                (__attribute__((address_space(3))) v4i*)(&sAlo[buf][loff]), 0, 0);
            __builtin_amdgcn_global_load_async_to_lds_b128(
                (__attribute__((address_space(1))) v4i*)(Bhi + gBoff),
                (__attribute__((address_space(3))) v4i*)(&sBhi[buf][loff]), 0, 0);
            __builtin_amdgcn_global_load_async_to_lds_b128(
                (__attribute__((address_space(1))) v4i*)(Blo + gBoff),
                (__attribute__((address_space(3))) v4i*)(&sBlo[buf][loff]), 0, 0);
#else
            *(v4i*)(&sAhi[buf][loff]) = *(const v4i*)(Ahi + gAoff);
            *(v4i*)(&sAlo[buf][loff]) = *(const v4i*)(Alo + gAoff);
            *(v4i*)(&sBhi[buf][loff]) = *(const v4i*)(Bhi + gBoff);
            *(v4i*)(&sBlo[buf][loff]) = *(const v4i*)(Blo + gBoff);
#endif
        }
    };

    v8f acc[MT][NT] = {};

    issue_tile(0, 0);
#if USE_ASYNC_LDS
    __builtin_amdgcn_s_wait_asynccnt(0);
#endif
    __syncthreads();

    for (int t = 0; t < NK; ++t) {
        const int cur = t & 1;
        if (t + 1 < NK) issue_tile((t + 1) * BK, cur ^ 1);

        // ---- A fragments: lane holds row (l%16); K halves at lg*8, 16+lg*8 ----
        v16bf aHi[MT], aLo[MT];
        #pragma unroll
        for (int i = 0; i < MT; ++i) {
            const int base = (waveM * WM + i * 16 + l16) * BK + lg * 8;
            const v8bf h0 = *(const v8bf*)(&sAhi[cur][base]);
            const v8bf h1 = *(const v8bf*)(&sAhi[cur][base + 16]);
            const v8bf l0 = *(const v8bf*)(&sAlo[cur][base]);
            const v8bf l1 = *(const v8bf*)(&sAlo[cur][base + 16]);
            #pragma unroll
            for (int e = 0; e < 8; ++e) {
                aHi[i][e] = h0[e]; aHi[i][e + 8] = h1[e];
                aLo[i][e] = l0[e]; aLo[i][e + 8] = l1[e];
            }
        }

        // ---- B fragments + WMMA: lane holds column (l%16); K halves at lg*8, 16+lg*8 ----
        #pragma unroll
        for (int j = 0; j < NT; ++j) {
            const int base = (waveN * WN + j * 16 + l16) * BK + lg * 8;
            const v8bf h0 = *(const v8bf*)(&sBhi[cur][base]);
            const v8bf h1 = *(const v8bf*)(&sBhi[cur][base + 16]);
            const v8bf l0 = *(const v8bf*)(&sBlo[cur][base]);
            const v8bf l1 = *(const v8bf*)(&sBlo[cur][base + 16]);
            v16bf bHi, bLo;
            #pragma unroll
            for (int e = 0; e < 8; ++e) {
                bHi[e] = h0[e]; bHi[e + 8] = h1[e];
                bLo[e] = l0[e]; bLo[e + 8] = l1[e];
            }
            #pragma unroll
            for (int i = 0; i < MT; ++i) {
                acc[i][j] = __builtin_amdgcn_wmma_f32_16x16x32_bf16(
                    false, aHi[i], false, bHi, (short)0, acc[i][j], false, false);
                acc[i][j] = __builtin_amdgcn_wmma_f32_16x16x32_bf16(
                    false, aHi[i], false, bLo, (short)0, acc[i][j], false, false);
                acc[i][j] = __builtin_amdgcn_wmma_f32_16x16x32_bf16(
                    false, aLo[i], false, bHi, (short)0, acc[i][j], false, false);
            }
        }

#if USE_ASYNC_LDS
        __builtin_amdgcn_s_wait_asynccnt(0);
#endif
        __syncthreads();
    }

    // ---- epilogue: C/D layout — VGPR r holds M = r + 8*lg, column N = l%16 ----
    #pragma unroll
    for (int j = 0; j < NT; ++j) {
        const int col = nBlock + waveN * WN + j * 16 + l16;
        const float bv = bias[col];
        #pragma unroll
        for (int i = 0; i < MT; ++i) {
            const int rbase = mBlock + waveM * WM + i * 16 + lg * 8;
            #pragma unroll
            for (int r = 0; r < 8; ++r) {
                out[(size_t)(rbase + r) * NTOT + col] = acc[i][j][r] + bv;
            }
        }
    }
}

// ---------------------------------------------------------------------------
// Log-space sequential scan, one thread per (b,h) channel, coalesced over h.
//   softplus(x) = max(x,0) + log1p(exp(-|x|))
//   log g(v)    = v>=0 ? log(v+0.5) : -softplus(-v)
//   log_h_t     = logaddexp(-softplus(k) + log_h_{t-1},
//                           -softplus(-k) + log g(pre))
// ---------------------------------------------------------------------------
__global__ __launch_bounds__(256)
void mingru_scan(const float* __restrict__ Kpre,
                 const float* __restrict__ Ppre,
                 float* __restrict__ H)
{
    const int tid = blockIdx.x * blockDim.x + threadIdx.x;  // 0..B*H-1
    const int b = tid / DH;
    const int h = tid - b * DH;
    const size_t base = (size_t)b * SEQ * DH + h;

    float log_h = -0.6931471805599453f;   // log(0.5) = log g(0)

    for (int t = 0; t < SEQ; ++t) {
        const size_t idx = base + (size_t)t * DH;
        const float kv = Kpre[idx];
        const float pv = Ppre[idx];

        const float sp_k  = fmaxf(kv, 0.0f) + log1pf(__expf(-fabsf(kv))); // softplus(k)
        const float log_c = -sp_k;            // log(1-z)
        const float log_z = kv - sp_k;        // -softplus(-k)

        const float log_th = (pv >= 0.0f)
            ? __logf(pv + 0.5f)
            : -(fmaxf(-pv, 0.0f) + log1pf(__expf(-fabsf(pv))));  // -softplus(-pv)

        const float lb = log_z + log_th;
        const float a  = log_c + log_h;
        const float m  = fmaxf(a, lb);
        log_h = m + log1pf(__expf(-fabsf(a - lb)));

        H[idx] = __expf(log_h);
    }
}

// ---------------------------------------------------------------------------
// Launch.
// Workspace layout (bytes):
//   [0,128M)        k            f32 (MTOT x NTOT)
//   [128M,256M)     pre          f32
//   [256M,320M)     x_hi         bf16 (MTOT x KTOT)
//   [320M,384M)     x_lo         bf16
//   [384M,...)      Wz_hi, Wz_lo, Wh_hi, Wh_lo   bf16 (NTOT x KTOT each, 2MB)
// Total ~392 MB; harness-provided d_ws assumed sized for the problem.
// ---------------------------------------------------------------------------
extern "C" void kernel_launch(void* const* d_in, const int* in_sizes, int n_in,
                              void* d_out, int out_size, void* d_ws, size_t ws_size,
                              hipStream_t stream)
{
    (void)in_sizes; (void)n_in; (void)out_size; (void)ws_size;

    const float* x  = (const float*)d_in[0];
    const float* Wz = (const float*)d_in[1];
    const float* bz = (const float*)d_in[2];
    const float* Wh = (const float*)d_in[3];
    const float* bh = (const float*)d_in[4];
    float* out = (float*)d_out;

    const size_t SZ_KP  = (size_t)MTOT * NTOT * sizeof(float);    // 128 MB
    const size_t SZ_X16 = (size_t)MTOT * KTOT * sizeof(__bf16);   // 64 MB
    const size_t SZ_W16 = (size_t)NTOT * KTOT * sizeof(__bf16);   // 2 MB

    char* ws = (char*)d_ws;
    float*  wsK  = (float*)(ws);
    float*  wsP  = (float*)(ws + SZ_KP);
    __bf16* xhi  = (__bf16*)(ws + 2 * SZ_KP);
    __bf16* xlo  = (__bf16*)(ws + 2 * SZ_KP + SZ_X16);
    __bf16* wzhi = (__bf16*)(ws + 2 * SZ_KP + 2 * SZ_X16);
    __bf16* wzlo = (__bf16*)(ws + 2 * SZ_KP + 2 * SZ_X16 + SZ_W16);
    __bf16* whhi = (__bf16*)(ws + 2 * SZ_KP + 2 * SZ_X16 + 2 * SZ_W16);
    __bf16* whlo = (__bf16*)(ws + 2 * SZ_KP + 2 * SZ_X16 + 3 * SZ_W16);

    // pre-split f32 -> (hi,lo) bf16
    split_bf16<<<(MTOT * KTOT / 4) / 256, 256, 0, stream>>>(x,  xhi,  xlo);
    split_bf16<<<(NTOT * KTOT / 4) / 256, 256, 0, stream>>>(Wz, wzhi, wzlo);
    split_bf16<<<(NTOT * KTOT / 4) / 256, 256, 0, stream>>>(Wh, whhi, whlo);

    dim3 grid(MTOT / BM, NTOT / BN);   // 256 x 8
    mingru_gemm_bf16x3<<<grid, 256, 0, stream>>>(xhi, xlo, wzhi, wzlo, bz, wsK);
    mingru_gemm_bf16x3<<<grid, 256, 0, stream>>>(xhi, xlo, whhi, whlo, bh, wsP);

    mingru_scan<<<(BATCH * DH) / 256, 256, 0, stream>>>(wsK, wsP, out);
}
